// MLPBlock_66503273611399
// MI455X (gfx1250) — matrix operand order repaired and checked by reading, not compiled, not run on hardware
//
#include <hip/hip_runtime.h>
#include <hip/hip_bf16.h>
#include <cstdint>

// ---------------------------------------------------------------------------
// MoE MLP block for MI455X (gfx1250, wave32, WMMA).
// Routed compute: 206 GFLOP; fp32 weights 805MB (~35us @ 23.3TB/s, streamed
// ~once due to z-slow tile order + 192MB L2). bf16 WMMA (16x16x32), f32 acc.
// 32x32 register tile per wave (4 acc), 64x128 block tile, explicit ping-pong
// double-buffered LDS (ds_store_b128 with immediate offsets), async-to-LDS
// staging for bf16 A tiles (ASYNCcnt). Steady-state K loop is branch-free
// (tail peeled) so accumulators stay pinned (no v_mov chains / phi copies).
// GEMM1 stages B with even/odd column separation so swiglu needs no shuffles.
// ---------------------------------------------------------------------------

typedef __bf16 bf16_t;
typedef bf16_t v16bf __attribute__((ext_vector_type(16)));
typedef bf16_t v8bf  __attribute__((ext_vector_type(8)));
typedef float  v8f   __attribute__((ext_vector_type(8)));

constexpr int T = 2048;
constexpr int H = 2048;
constexpr int I = 2048;
constexpr int E = 16;
constexpr int TOPK = 4;
constexpr float EPS   = 1e-5f;
constexpr float ALPHA = 1.702f;
constexpr float LIMIT = 7.0f;

// LDS row stride in bf16 elements: 40 halves = 80B => 16B-aligned fragment
// chunks land on distinct banks for 16 consecutive rows.
constexpr int SA = 40;
constexpr int SB = 40;

#define WMMA_BF16(a, b, c) \
  __builtin_amdgcn_wmma_f32_16x16x32_bf16(false, (a), false, (b), (short)0, (c), false, false)

// --- CDNA5 async copy: global -> LDS (tracked by ASYNCcnt) ------------------
__device__ __forceinline__ void async_copy_b128(uint32_t lds_addr, const void* gsrc) {
  // VDST = per-lane LDS byte address, VADDR = per-lane 64-bit global address.
  asm volatile("global_load_async_to_lds_b128 %0, %1, off"
               :: "v"(lds_addr), "v"(gsrc)
               : "memory");
}
__device__ __forceinline__ void wait_async0() {
#if __has_builtin(__builtin_amdgcn_s_wait_asynccnt)
  __builtin_amdgcn_s_wait_asynccnt(0);
#else
  asm volatile("s_wait_asynccnt 0x0" ::: "memory");
#endif
}
__device__ __forceinline__ uint32_t lds_addr_of(const void* p) {
  // Low 32 bits of a flat pointer into LDS == wave-relative LDS byte address.
  return (uint32_t)(uintptr_t)p;
}

// --- fragment loaders (contiguous k = 16*(lane/16) + s for A and B, wave32) --------------------
__device__ __forceinline__ v16bf load_a_frag(const bf16_t* sA_, int mbase, int lane) {
  const int lr = lane & 15, hl = lane >> 4;
  const bf16_t* rp = sA_ + (mbase + lr) * SA;
  v8bf c0 = *(const v8bf*)(rp + hl * 16);
  v8bf c1 = *(const v8bf*)(rp + hl * 16 + 8);
  v16bf a;
#pragma unroll
  for (int i = 0; i < 8; ++i) { a[i] = c0[i]; a[8 + i] = c1[i]; }
  return a;
}
__device__ __forceinline__ v16bf load_b_frag(const bf16_t* sB_, int nbase, int lane) {
  const int lr = lane & 15, hl = lane >> 4;
  const bf16_t* rp = sB_ + (nbase + lr) * SB + hl * 16;
  v8bf c0 = *(const v8bf*)(rp);
  v8bf c1 = *(const v8bf*)(rp + 8);
  v16bf b;
#pragma unroll
  for (int i = 0; i < 8; ++i) { b[i] = c0[i]; b[8 + i] = c1[i]; }
  return b;
}

__device__ __forceinline__ void stage_load(float4 f[4], uint32_t ldsA_addr,
                                           const bf16_t* asrc, const float* bsrc) {
  async_copy_b128(ldsA_addr, asrc);
  f[0] = *(const float4*)(bsrc + 0);
  f[1] = *(const float4*)(bsrc + 4);
  f[2] = *(const float4*)(bsrc + 8);
  f[3] = *(const float4*)(bsrc + 12);
  __builtin_prefetch(bsrc + 128, 0, 1);
}
__device__ __forceinline__ void stage_store(bf16_t* dst, const float4* f) {
  v8bf v0, v1;
  v0[0] = (bf16_t)f[0].x; v0[1] = (bf16_t)f[0].y; v0[2] = (bf16_t)f[0].z; v0[3] = (bf16_t)f[0].w;
  v0[4] = (bf16_t)f[1].x; v0[5] = (bf16_t)f[1].y; v0[6] = (bf16_t)f[1].z; v0[7] = (bf16_t)f[1].w;
  v1[0] = (bf16_t)f[2].x; v1[1] = (bf16_t)f[2].y; v1[2] = (bf16_t)f[2].z; v1[3] = (bf16_t)f[2].w;
  v1[4] = (bf16_t)f[3].x; v1[5] = (bf16_t)f[3].y; v1[6] = (bf16_t)f[3].z; v1[7] = (bf16_t)f[3].w;
  *(v8bf*)(dst) = v0;
  *(v8bf*)(dst + 8) = v1;
}

__device__ __forceinline__ void gemm_compute(const bf16_t* sAbuf, const bf16_t* sBbuf,
                                             int wm, int wn, int lane, v8f (&acc)[2][2]) {
  v16bf a0 = load_a_frag(sAbuf, wm * 32, lane);
  v16bf a1 = load_a_frag(sAbuf, wm * 32 + 16, lane);
  v16bf b0 = load_b_frag(sBbuf, wn * 32, lane);
  v16bf b1 = load_b_frag(sBbuf, wn * 32 + 16, lane);
  acc[0][0] = WMMA_BF16(a0, b0, acc[0][0]);
  acc[0][1] = WMMA_BF16(a0, b1, acc[0][1]);
  acc[1][0] = WMMA_BF16(a1, b0, acc[1][0]);
  acc[1][1] = WMMA_BF16(a1, b1, acc[1][1]);
}

// ---------------------------------------------------------------------------
// Kernel 1: RMSNorm + router (gate GEMV, top-4, softmax, slot assignment)
// ---------------------------------------------------------------------------
__global__ __launch_bounds__(256) void router_kernel(
    const float* __restrict__ x, const float* __restrict__ scale,
    const float* __restrict__ gk, const float* __restrict__ gb,
    bf16_t* __restrict__ tn, int* __restrict__ cnt, int* __restrict__ etok,
    float* __restrict__ slot_w, int* __restrict__ tke, int* __restrict__ tkp) {
  const int t = blockIdx.x, tid = threadIdx.x;
  __shared__ float pbuf[16 * 256];
  __shared__ float red[256];
  __shared__ float gsh[16];

  float xv[8];
#pragma unroll
  for (int i = 0; i < 8; ++i) xv[i] = x[(size_t)t * H + tid * 8 + i];
  float ss = 0.f;
#pragma unroll
  for (int i = 0; i < 8; ++i) ss += xv[i] * xv[i];
  red[tid] = ss;
  __syncthreads();
  for (int s = 128; s > 0; s >>= 1) {
    if (tid < s) red[tid] += red[tid + s];
    __syncthreads();
  }
  const float invr = rsqrtf(red[0] * (1.0f / H) + EPS);

  float part[16];
#pragma unroll
  for (int e = 0; e < 16; ++e) part[e] = 0.f;
#pragma unroll
  for (int i = 0; i < 8; ++i) {
    const int h = tid * 8 + i;
    const float tv = xv[i] * invr * scale[h];
    tn[(size_t)t * H + h] = (bf16_t)tv;
    const float* gr = gk + h * 16;
#pragma unroll
    for (int e = 0; e < 16; ++e) part[e] += tv * gr[e];
  }
#pragma unroll
  for (int e = 0; e < 16; ++e) pbuf[e * 256 + tid] = part[e];
  __syncthreads();
  if (tid < 16) {
    float s = 0.f;
    for (int j = 0; j < 256; ++j) s += pbuf[tid * 256 + j];
    gsh[tid] = s + gb[tid];
  }
  __syncthreads();
  if (tid == 0) {
    float lg[16];
#pragma unroll
    for (int e = 0; e < 16; ++e) lg[e] = gsh[e];
    int idx[TOPK]; float val[TOPK]; bool used[16];
#pragma unroll
    for (int e = 0; e < 16; ++e) used[e] = false;
    for (int j = 0; j < TOPK; ++j) {
      int b = 0; float bv = -3.0e38f;
      for (int e = 0; e < 16; ++e)
        if (!used[e] && lg[e] > bv) { bv = lg[e]; b = e; }
      used[b] = true; idx[j] = b; val[j] = bv;
    }
    const float m = val[0];
    float ex[TOPK], den = 0.f;
    for (int j = 0; j < TOPK; ++j) { ex[j] = expf(val[j] - m); den += ex[j]; }
    for (int j = 0; j < TOPK; ++j) {
      const int e = idx[j];
      const float wj = ex[j] / den;
      const int pos = atomicAdd(&cnt[e], 1);
      etok[e * T + pos] = t;
      slot_w[e * T + pos] = wj;
      tke[t * TOPK + j] = e;
      tkp[t * TOPK + j] = pos;
    }
  }
}

// Kernel 2: exclusive prefix over 16 expert counts
__global__ void prefix_kernel(const int* __restrict__ cnt, int* __restrict__ base) {
  if (threadIdx.x == 0) {
    int s = 0;
    for (int e = 0; e < E; ++e) { base[e] = s; s += cnt[e]; }
  }
}

// ---------------------------------------------------------------------------
// Kernel 3: grouped GEMM1 (M=n_e, N=2I, K=H) + swiglu -> act (bf16)
// ---------------------------------------------------------------------------
__global__ __launch_bounds__(256) void gemm1_swiglu_kernel(
    const bf16_t* __restrict__ tn, const float* __restrict__ w1,
    const float* __restrict__ b1, const int* __restrict__ cnt,
    const int* __restrict__ base, const int* __restrict__ etok,
    bf16_t* __restrict__ act) {
  const int e = blockIdx.x;
  const int n = cnt[e];
  const int rowbase = blockIdx.y * 64;
  if (rowbase >= n) return;
  const int c0 = blockIdx.z * 128;  // output column tile in [0, 2I)

  __shared__ __align__(16) bf16_t sA[2][64 * SA];
  __shared__ __align__(16) bf16_t sB[2][128 * SB];
  __shared__ int rows[64];

  const int tid = threadIdx.x;
  if (tid < 64) {
    const int r = rowbase + tid;
    rows[tid] = etok[e * T + (r < n ? r : rowbase)];
  }
  __syncthreads();

  // A staging: 64 rows x 32 halves, one async b128 per thread.
  const int arow = tid >> 2;
  const int achunk = (tid & 3) * 8;
  const bf16_t* asrc = tn + (size_t)rows[arow] * H + achunk;
  const uint32_t ldsA0 = lds_addr_of(&sA[0][arow * SA + achunk]);
  const uint32_t ldsA1 = lds_addr_of(&sA[1][arow * SA + achunk]);

  // B staging: 128 rows x 32 floats (16 per thread), even/odd row permutation
  // within each 32-row group: even cols -> local rows [0,16), odd -> [16,32).
  const int brow = tid >> 1;
  const int bkoff = (tid & 1) * 16;
  const int bgrp = brow >> 5, bw = brow & 31;
  const int browp = bgrp * 32 + ((bw & 1) << 4) + (bw >> 1);
  const float* bsrc = w1 + ((size_t)e * (2 * I) + (c0 + brow)) * H + bkoff;
  bf16_t* const ldsB0 = &sB[0][browp * SB + bkoff];
  bf16_t* const ldsB1 = &sB[1][browp * SB + bkoff];

  const int lane = tid & 31, w = tid >> 5;
  const int wm = w >> 2, wn = w & 3;

  v8f acc[2][2] = {};

  {
    float4 f[4];
    stage_load(f, ldsA0, asrc, bsrc);
    stage_store(ldsB0, f);
  }
  wait_async0();
  __syncthreads();

  // steady state: branch-free body, tail peeled
#pragma unroll 1
  for (int k0 = 0; k0 < H - 64; k0 += 64) {
    {
      float4 f[4];
      stage_load(f, ldsA1, asrc + k0 + 32, bsrc + k0 + 32);
      gemm_compute(sA[0], sB[0], wm, wn, lane, acc);
      stage_store(ldsB1, f);
    }
    wait_async0();
    __syncthreads();
    {
      float4 f[4];
      stage_load(f, ldsA0, asrc + k0 + 64, bsrc + k0 + 64);
      gemm_compute(sA[1], sB[1], wm, wn, lane, acc);
      stage_store(ldsB0, f);
    }
    wait_async0();
    __syncthreads();
  }
  {
    float4 f[4];
    stage_load(f, ldsA1, asrc + H - 32, bsrc + H - 32);
    gemm_compute(sA[0], sB[0], wm, wn, lane, acc);
    stage_store(ldsB1, f);
  }
  wait_async0();
  __syncthreads();
  gemm_compute(sA[1], sB[1], wm, wn, lane, acc);

  // shuffle-free swiglu epilogue: glu = acc[mi][0], lin = acc[mi][1]
  const int lr = lane & 15, hl = lane >> 4;
  const int gbase = base[e];
  const int oeven = c0 + wn * 32 + 2 * lr;        // even output column
  const float be = b1[e * (2 * I) + oeven];
  const float bo = b1[e * (2 * I) + oeven + 1];
  const int icol = (c0 >> 1) + wn * 16 + lr;      // act column in [0, I)
#pragma unroll
  for (int mi = 0; mi < 2; ++mi) {
#pragma unroll
    for (int r = 0; r < 8; ++r) {
      const int m = wm * 32 + mi * 16 + r + 8 * hl;
      if (rowbase + m < n) {
        const float glu = fminf(acc[mi][0][r] + be, LIMIT);
        const float lin = fminf(fmaxf(acc[mi][1][r] + bo, -LIMIT), LIMIT);
        const float sg = 1.0f / (1.0f + expf(-ALPHA * glu));
        act[(size_t)(gbase + rowbase + m) * I + icol] =
            (bf16_t)(glu * sg * (lin + 1.0f));
      }
    }
  }
}

// ---------------------------------------------------------------------------
// Kernel 4: grouped GEMM2 (M=n_e, N=H, K=I); epilogue folds bias + routing
// weight into per-slot f32 rows (deterministic combine, no float atomics).
// ---------------------------------------------------------------------------
__global__ __launch_bounds__(256) void gemm2_kernel(
    const bf16_t* __restrict__ act, const float* __restrict__ w2,
    const float* __restrict__ b2, const int* __restrict__ cnt,
    const int* __restrict__ base, const float* __restrict__ slot_w,
    float* __restrict__ h2b) {
  const int e = blockIdx.x;
  const int n = cnt[e];
  const int rowbase = blockIdx.y * 64;
  if (rowbase >= n) return;
  const int c0 = blockIdx.z * 128;  // output column tile in [0, H)

  __shared__ __align__(16) bf16_t sA[2][64 * SA];
  __shared__ __align__(16) bf16_t sB[2][128 * SB];
  __shared__ float wbuf[64];

  const int tid = threadIdx.x;
  const int gbase = base[e];
  if (tid < 64) {
    const int r = rowbase + tid;
    wbuf[tid] = (r < n) ? slot_w[e * T + r] : 0.0f;
  }

  const int arow = tid >> 2;
  const int achunk = (tid & 3) * 8;
  const int ar = rowbase + arow;
  const bf16_t* asrc = act + (size_t)(gbase + (ar < n ? ar : rowbase)) * I + achunk;
  const uint32_t ldsA0 = lds_addr_of(&sA[0][arow * SA + achunk]);
  const uint32_t ldsA1 = lds_addr_of(&sA[1][arow * SA + achunk]);

  const int brow = tid >> 1;
  const int bkoff = (tid & 1) * 16;
  const float* bsrc = w2 + ((size_t)e * H + (c0 + brow)) * I + bkoff;
  bf16_t* const ldsB0 = &sB[0][brow * SB + bkoff];
  bf16_t* const ldsB1 = &sB[1][brow * SB + bkoff];

  const int lane = tid & 31, w = tid >> 5;
  const int wm = w >> 2, wn = w & 3;

  v8f acc[2][2] = {};

  {
    float4 f[4];
    stage_load(f, ldsA0, asrc, bsrc);
    stage_store(ldsB0, f);
  }
  wait_async0();
  __syncthreads();

#pragma unroll 1
  for (int k0 = 0; k0 < I - 64; k0 += 64) {
    {
      float4 f[4];
      stage_load(f, ldsA1, asrc + k0 + 32, bsrc + k0 + 32);
      gemm_compute(sA[0], sB[0], wm, wn, lane, acc);
      stage_store(ldsB1, f);
    }
    wait_async0();
    __syncthreads();
    {
      float4 f[4];
      stage_load(f, ldsA0, asrc + k0 + 64, bsrc + k0 + 64);
      gemm_compute(sA[1], sB[1], wm, wn, lane, acc);
      stage_store(ldsB0, f);
    }
    wait_async0();
    __syncthreads();
  }
  {
    float4 f[4];
    stage_load(f, ldsA1, asrc + I - 32, bsrc + I - 32);
    gemm_compute(sA[0], sB[0], wm, wn, lane, acc);
    stage_store(ldsB1, f);
  }
  wait_async0();
  __syncthreads();
  gemm_compute(sA[1], sB[1], wm, wn, lane, acc);

  const int lr = lane & 15, hl = lane >> 4;
#pragma unroll
  for (int ni = 0; ni < 2; ++ni) {
    const int col = c0 + wn * 32 + ni * 16 + lr;  // [0, H)
    const float bias = b2[e * H + col];
#pragma unroll
    for (int mi = 0; mi < 2; ++mi) {
#pragma unroll
      for (int r = 0; r < 8; ++r) {
        const int m = wm * 32 + mi * 16 + r + 8 * hl;
        if (rowbase + m < n)
          h2b[(size_t)(gbase + rowbase + m) * H + col] =
              (acc[mi][ni][r] + bias) * wbuf[m];
      }
    }
  }
}

// ---------------------------------------------------------------------------
// Kernel 5: out[t] = x[t] + sum_j h2[slot(t,j)] (fixed j order, deterministic)
// ---------------------------------------------------------------------------
__global__ __launch_bounds__(256) void combine_kernel(
    const float* __restrict__ x, const float* __restrict__ h2b,
    const int* __restrict__ tke, const int* __restrict__ tkp,
    const int* __restrict__ base, float* __restrict__ out) {
  const int t = blockIdx.x, tid = threadIdx.x;
  int g[TOPK];
#pragma unroll
  for (int j = 0; j < TOPK; ++j)
    g[j] = base[tke[t * TOPK + j]] + tkp[t * TOPK + j];
#pragma unroll
  for (int i = 0; i < 8; ++i) {
    const int c = i * 256 + tid;
    float s = x[(size_t)t * H + c];
#pragma unroll
    for (int j = 0; j < TOPK; ++j) s += h2b[(size_t)g[j] * H + c];
    out[(size_t)t * H + c] = s;
  }
}

// ---------------------------------------------------------------------------
extern "C" void kernel_launch(void* const* d_in, const int* in_sizes, int n_in,
                              void* d_out, int out_size, void* d_ws, size_t ws_size,
                              hipStream_t stream) {
  (void)in_sizes; (void)n_in; (void)out_size; (void)ws_size;
  const float* x  = (const float*)d_in[0];
  const float* ns = (const float*)d_in[1];
  const float* gk = (const float*)d_in[2];
  const float* gb = (const float*)d_in[3];
  const float* w1 = (const float*)d_in[4];
  const float* b1 = (const float*)d_in[5];
  const float* w2 = (const float*)d_in[6];
  const float* b2 = (const float*)d_in[7];
  float* out = (float*)d_out;

  char* p = (char*)d_ws;
  auto take = [&](size_t bytes) -> char* {
    char* q = p;
    p += (bytes + 255) & ~(size_t)255;
    return q;
  };
  bf16_t* tn   = (bf16_t*)take((size_t)T * H * sizeof(bf16_t));          // 8 MB
  bf16_t* act  = (bf16_t*)take((size_t)T * TOPK * I * sizeof(bf16_t));   // 32 MB
  float*  h2b  = (float*) take((size_t)T * TOPK * H * sizeof(float));    // 64 MB
  int*    etok = (int*)   take((size_t)E * T * sizeof(int));
  float*  sw   = (float*) take((size_t)E * T * sizeof(float));
  int*    cnt  = (int*)   take(E * sizeof(int));
  int*    base = (int*)   take(E * sizeof(int));
  int*    tke  = (int*)   take((size_t)T * TOPK * sizeof(int));
  int*    tkp  = (int*)   take((size_t)T * TOPK * sizeof(int));

  hipMemsetAsync(cnt, 0, E * sizeof(int), stream);
  router_kernel<<<T, 256, 0, stream>>>(x, ns, gk, gb, tn, cnt, etok, sw, tke, tkp);
  prefix_kernel<<<1, 32, 0, stream>>>(cnt, base);
  gemm1_swiglu_kernel<<<dim3(E, T / 64, (2 * I) / 128), 256, 0, stream>>>(
      tn, w1, b1, cnt, base, etok, act);
  gemm2_kernel<<<dim3(E, T / 64, H / 128), 256, 0, stream>>>(
      act, w2, b2, cnt, base, sw, h2b);
  combine_kernel<<<T, 256, 0, stream>>>(x, h2b, tke, tkp, base, out);
}
